// GMMSVD_88854283420168
// MI455X (gfx1250) — matrix operand order repaired and checked by reading, not compiled, not run on hardware
//
#include <hip/hip_runtime.h>
#include <hip/hip_bf16.h>
#include <math.h>

typedef __attribute__((ext_vector_type(16))) _Float16 v16h;
typedef __attribute__((ext_vector_type(8)))  _Float16 v8h;
typedef __attribute__((ext_vector_type(8)))  float    v8f;

namespace cfg {
constexpr int B  = 32;
constexpr int N  = 4096;   // points
constexpr int J  = 128;    // mixture components
constexpr int JO = 129;    // J + outlier column
constexpr int JP = 144;    // JO padded to multiple of 16 for WMMA
constexpr int C  = 256;    // descriptor channels
constexpr int CB = 4;      // C-tiles per wave in the desc_mu GEMM (16x64 strip)
constexpr int CSTR = 16656;            // per-batch cost stride (floats), 16B-aligned
constexpr float EPSV   = 1e-8f;
constexpr float SK_EPS = 0.01f;
constexpr int   SK_IT  = 30;
}

// ---------------------------------------------------------------------------
// 1) Fused softmax over J + outlier weight, emitted as f16 W^T (B, JP, N).
// ---------------------------------------------------------------------------
__global__ __launch_bounds__(256) void k_softmax_w(const float* __restrict__ lg,
                                                   const float* __restrict__ o,
                                                   _Float16* __restrict__ Wt)
{
    using namespace cfg;
    const int b = blockIdx.y;
    const int n = blockIdx.x * 256 + threadIdx.x;
    const float* col = lg + (size_t)b * J * N + n;
    float mx = -3.4e38f, se = 0.f;                 // online softmax
    for (int j = 0; j < J; ++j) {
        float x  = col[(size_t)j * N];
        float nm = fmaxf(mx, x);
        se = se * __expf(mx - nm) + __expf(x - nm);
        mx = nm;
    }
    const float ov    = o[(size_t)b * N + n];
    const float scale = ov / se;
    _Float16* wc = Wt + (size_t)b * JP * N + n;
    for (int j = 0; j < J; ++j) {
        float x = col[(size_t)j * N];
        wc[(size_t)j * N] = (_Float16)(__expf(x - mx) * scale);
    }
    wc[(size_t)J * N] = (_Float16)(1.0f - ov);
    for (int j = J + 1; j < JP; ++j)
        wc[(size_t)j * N] = (_Float16)0.0f;
}

// ---------------------------------------------------------------------------
// 2) Column sums of W -> pi and denom (f32).
// ---------------------------------------------------------------------------
__global__ __launch_bounds__(256) void k_colsum(const _Float16* __restrict__ Wt,
                                                float* __restrict__ denom,
                                                float* __restrict__ pi)
{
    using namespace cfg;
    __shared__ float r[256];
    const int j = blockIdx.x, b = blockIdx.y;
    const _Float16* row = Wt + ((size_t)b * JP + j) * N;
    float s = 0.f;
    for (int n = threadIdx.x; n < N; n += 256) s += (float)row[n];
    r[threadIdx.x] = s; __syncthreads();
    for (int st = 128; st > 0; st >>= 1) {
        if (threadIdx.x < st) r[threadIdx.x] += r[threadIdx.x + st];
        __syncthreads();
    }
    if (threadIdx.x == 0) {
        denom[(size_t)b * JP + j] = r[0] + EPSV;
        pi   [(size_t)b * JP + j] = r[0] / (float)N;
    }
}

// ---------------------------------------------------------------------------
// 3) mu[b][j][d] = sum_n Wt[b][j][n] * xyz[b][d][n] / denom
// ---------------------------------------------------------------------------
__global__ __launch_bounds__(256) void k_mu(const _Float16* __restrict__ Wt,
                                            const float* __restrict__ xyz,
                                            const float* __restrict__ denom,
                                            float* __restrict__ mu)
{
    using namespace cfg;
    __shared__ float r[256];
    const int j = blockIdx.x, b = blockIdx.y;
    const _Float16* row = Wt + ((size_t)b * JP + j) * N;
    const float* x0 = xyz + (size_t)b * 3 * N;
    float a[3] = {0.f, 0.f, 0.f};
    for (int n = threadIdx.x; n < N; n += 256) {
        float w = (float)row[n];
        a[0] += w * x0[n]; a[1] += w * x0[N + n]; a[2] += w * x0[2 * N + n];
    }
    const float d = denom[(size_t)b * JP + j];
    for (int dd = 0; dd < 3; ++dd) {
        r[threadIdx.x] = a[dd]; __syncthreads();
        for (int st = 128; st > 0; st >>= 1) {
            if (threadIdx.x < st) r[threadIdx.x] += r[threadIdx.x + st];
            __syncthreads();
        }
        if (threadIdx.x == 0) mu[((size_t)b * JP + j) * 3 + dd] = r[0] / d;
        __syncthreads();
    }
}

// ---------------------------------------------------------------------------
// 4) Transpose + downconvert descriptors: (B,N,C) f32 -> (B,C,N) f16.
// ---------------------------------------------------------------------------
__global__ __launch_bounds__(256) void k_transpose(const float* __restrict__ desc,
                                                   _Float16* __restrict__ descT)
{
    using namespace cfg;
    __shared__ float tile[32][33];
    const int b = blockIdx.z;
    const int n0 = blockIdx.x * 32, c0 = blockIdx.y * 32;
    const int tx = threadIdx.x, ty = threadIdx.y;
    #pragma unroll
    for (int i = 0; i < 4; ++i)
        tile[ty + i * 8][tx] = desc[((size_t)b * N + n0 + ty + i * 8) * C + c0 + tx];
    __syncthreads();
    #pragma unroll
    for (int i = 0; i < 4; ++i)
        descT[((size_t)b * C + c0 + ty + i * 8) * N + n0 + tx] =
            (_Float16)tile[tx][ty + i * 8];
}

// ---------------------------------------------------------------------------
// 5) desc_mu GEMM, register-blocked: one wave computes a 16(M) x 64(C) strip,
//    K=4096 in steps of 32. A tile is loaded once per K-step and reused by
//    4 back-to-back v_wmma_f32_16x16x32_f16 (4x less A traffic, 4x WMMA
//    density per issued load).
// ---------------------------------------------------------------------------
__global__ __launch_bounds__(256) void k_gemm_descmu(const _Float16* __restrict__ Aw,
                                                     const _Float16* __restrict__ Bw,
                                                     const float* __restrict__ denom,
                                                     float* __restrict__ P)
{
    using namespace cfg;
    const int wave = threadIdx.x >> 5;
    const int lane = threadIdx.x & 31;
    const int tilesPerB = (JP / 16) * (C / (16 * CB));   // 9 * 4 = 36
    const int gw = blockIdx.x * 8 + wave;
    const int b  = gw / tilesPerB;
    if (b >= B) return;                                   // wave-uniform
    const int t  = gw % tilesPerB;
    const int jt = t % (JP / 16);
    const int ct = t / (JP / 16);                         // 64-wide C strip
    const int half = lane >> 4, mrow = lane & 15;

    const _Float16* arow = Aw + ((size_t)b * JP + jt * 16 + mrow) * N;
    const _Float16* brow = Bw + ((size_t)b * C + ct * (16 * CB) + mrow) * N;

    v8f acc[CB] = {};
    for (int kb = 0; kb < N; kb += 32) {
        if ((kb & 255) == 0) {
            __builtin_prefetch(arow + kb + 256, 0, 1);
            __builtin_prefetch(brow + kb + 256, 0, 1);
        }
        // A (16x32): lane half 0 -> K 0..7 / 16..23 ; half 1 -> K 8..15 / 24..31
        v8h a0 = *(const v8h*)(arow + kb + half * 8);
        v8h a1 = *(const v8h*)(arow + kb + 16 + half * 8);
        v16h a;
        #pragma unroll
        for (int i = 0; i < 8; ++i) { a[i] = a0[i]; a[i + 8] = a1[i]; }
        #pragma unroll
        for (int cc = 0; cc < CB; ++cc) {
            // B (32x16): lane half 0 -> K 0..7 / 16..23 ; half 1 -> K 8..15 / 24..31
            const _Float16* bp = brow + (size_t)cc * 16 * N + kb + half * 8;
            v8h b0 = *(const v8h*)(bp);
            v8h b1 = *(const v8h*)(bp + 16);
            v16h bb;
            #pragma unroll
            for (int i = 0; i < 8; ++i) { bb[i] = b0[i]; bb[i + 8] = b1[i]; }
            acc[cc] = __builtin_amdgcn_wmma_f32_16x16x32_f16(false, a, false, bb,
                                                             (short)0, acc[cc],
                                                             false, false);
        }
    }
    #pragma unroll
    for (int cc = 0; cc < CB; ++cc)
        #pragma unroll
        for (int g = 0; g < 8; ++g) {                     // C/D: M = g + 8*half
            const int m = jt * 16 + g + half * 8;
            const int c = ct * (16 * CB) + cc * 16 + mrow;
            P[((size_t)b * JP + m) * C + c] = acc[cc][g] / denom[(size_t)b * JP + m];
        }
}

// ---------------------------------------------------------------------------
// 6) Swap last channel between src/tgt, L2-normalize rows, emit f16 (B,JP,C).
// ---------------------------------------------------------------------------
__global__ __launch_bounds__(256) void k_swap_norm(const float* __restrict__ Pself,
                                                   const float* __restrict__ Pother,
                                                   _Float16* __restrict__ Nout)
{
    using namespace cfg;
    __shared__ float r[256];
    const int j = blockIdx.x, b = blockIdx.y, c = threadIdx.x;
    float v = Pself[((size_t)b * JP + j) * C + c];
    if (c == C - 1) v = Pother[((size_t)b * JP + j) * C + c];
    r[c] = v * v; __syncthreads();
    for (int st = 128; st > 0; st >>= 1) {
        if (c < st) r[c] += r[c + st];
        __syncthreads();
    }
    const float inv = 1.0f / (sqrtf(r[0]) + EPSV);
    Nout[((size_t)b * JP + j) * C + c] = (_Float16)(v * inv);
}

// ---------------------------------------------------------------------------
// 7) Similarity GEMM: cost[b][m][n] = 1 - <srcN[m,:], tgtN[n,:]> (K=256, WMMA).
// ---------------------------------------------------------------------------
__global__ __launch_bounds__(256) void k_gemm_sim(const _Float16* __restrict__ NS,
                                                  const _Float16* __restrict__ NT,
                                                  float* __restrict__ cost)
{
    using namespace cfg;
    const int wave = threadIdx.x >> 5;
    const int lane = threadIdx.x & 31;
    const int tilesPerB = (JP / 16) * (JP / 16);          // 81
    const int gw = blockIdx.x * 8 + wave;
    const int b  = gw / tilesPerB;
    if (b >= B) return;                                   // wave-uniform
    const int t  = gw % tilesPerB;
    const int mt = t % (JP / 16);
    const int nt = t / (JP / 16);
    const int half = lane >> 4, mrow = lane & 15;

    const _Float16* arow = NS + ((size_t)b * JP + mt * 16 + mrow) * C;
    const _Float16* brow = NT + ((size_t)b * JP + nt * 16 + mrow) * C;

    v8f acc = {};
    #pragma unroll
    for (int kb = 0; kb < C; kb += 32) {
        v8h a0 = *(const v8h*)(arow + kb + half * 8);
        v8h a1 = *(const v8h*)(arow + kb + 16 + half * 8);
        v8h b0 = *(const v8h*)(brow + kb + half * 8);
        v8h b1 = *(const v8h*)(brow + kb + 16 + half * 8);
        v16h a, bb;
        #pragma unroll
        for (int i = 0; i < 8; ++i) { a[i] = a0[i]; a[i + 8] = a1[i]; bb[i] = b0[i]; bb[i + 8] = b1[i]; }
        acc = __builtin_amdgcn_wmma_f32_16x16x32_f16(false, a, false, bb,
                                                     (short)0, acc, false, false);
    }
    #pragma unroll
    for (int g = 0; g < 8; ++g) {
        const int m = mt * 16 + g + half * 8;
        const int n = nt * 16 + mrow;
        if (m < JO && n < JO)
            cost[(size_t)b * CSTR + m * JO + n] = 1.0f - acc[g];
    }
}

// ---------------------------------------------------------------------------
// 8) Sinkhorn (30 iters) entirely in LDS. The 65 KB cost matrix is staged
//    global->LDS with GLOBAL_LOAD_ASYNC_TO_LDS_B128 (ASYNCcnt), then the
//    iterations run barriered in one workgroup per batch.
// ---------------------------------------------------------------------------
__global__ __launch_bounds__(256) void k_sinkhorn(const float* __restrict__ cost,
                                                  const float* __restrict__ piS,
                                                  const float* __restrict__ piT,
                                                  const float* __restrict__ muT,
                                                  float* __restrict__ outScores,
                                                  float* __restrict__ outCorrT,
                                                  float* __restrict__ weightW,
                                                  float* __restrict__ corrW)
{
    using namespace cfg;
    extern __shared__ float smem[];
    float* Cm  = smem;                 // CSTR floats (129x129 + pad), at LDS offset 0
    float* su  = Cm  + CSTR;           // 129
    float* sv  = su  + JO;             // 129
    float* lp  = sv  + JO;             // 129
    float* lq  = lp  + JO;             // 129
    float* tmu = lq  + JO;             // 128*3
    const int b = blockIdx.x, tid = threadIdx.x;

    // Async DMA of the padded cost row block into LDS (16B chunks, aligned).
    {
        const float* gsrc = cost + (size_t)b * CSTR;
        const unsigned ldsBase = (unsigned)(uintptr_t)(void*)Cm;  // low 32 bits of the pointer give the LDS address
        for (int idx = tid; idx < CSTR / 4; idx += 256) {
            const unsigned ldsOff = ldsBase + (unsigned)idx * 16u;
            const float* g = gsrc + (size_t)idx * 4;
            asm volatile("global_load_async_to_lds_b128 %0, %1, off"
                         :: "v"(ldsOff), "v"(g) : "memory");
        }
        asm volatile("s_wait_asynccnt 0x0" ::: "memory");
    }
    if (tid < JO) {
        su[tid] = 0.f; sv[tid] = 0.f;
        lp[tid] = __logf(piS[(size_t)b * JP + tid] + EPSV);
        lq[tid] = __logf(piT[(size_t)b * JP + tid] + EPSV);
    }
    for (int idx = tid; idx < J * 3; idx += 256)
        tmu[idx] = muT[((size_t)b * JP + idx / 3) * 3 + (idx % 3)];
    __syncthreads();

    for (int it = 0; it < SK_IT; ++it) {
        if (tid < JO) {                         // row (u) update
            const int i = tid; const float ui = su[i];
            float mx = -3.4e38f;
            for (int k = 0; k < JO; ++k) mx = fmaxf(mx, -Cm[i * JO + k] + ui + sv[k]);
            float se = 0.f;
            for (int k = 0; k < JO; ++k)
                se += __expf(((-Cm[i * JO + k] + ui + sv[k]) - mx) / SK_EPS);
            su[i] = SK_EPS * (lp[i] - (mx / SK_EPS + __logf(se))) + ui;
        }
        __syncthreads();
        if (tid < JO) {                         // column (v) update
            const int k = tid; const float vk = sv[k];
            float mx = -3.4e38f;
            for (int i = 0; i < JO; ++i) mx = fmaxf(mx, -Cm[i * JO + k] + su[i] + vk);
            float se = 0.f;
            for (int i = 0; i < JO; ++i)
                se += __expf(((-Cm[i * JO + k] + su[i] + vk) - mx) / SK_EPS);
            sv[k] = SK_EPS * (lq[k] - (mx / SK_EPS + __logf(se))) + vk;
        }
        __syncthreads();
    }
    // scores (only the 128x128 block is consumed) — transform in place
    for (int idx = tid; idx < J * J; idx += 256) {
        const int i = idx >> 7, k = idx & 127;
        Cm[i * JO + k] = __expf((-Cm[i * JO + k] + su[i] + sv[k]) / SK_EPS);
    }
    __syncthreads();
    if (tid < J) {
        const int i = tid;
        float rs = 0.f;
        for (int k = 0; k < J; ++k) rs += Cm[i * JO + k];
        const float inv = 1.0f / fmaxf(rs, 1e-4f);
        float wsum = 0.f, cx = 0.f, cy = 0.f, cz = 0.f;
        for (int k = 0; k < J; ++k) {
            const float s = Cm[i * JO + k] * inv;
            outScores[((size_t)b * J + i) * J + k] = s;
            wsum += s;
            cx += tmu[k * 3 + 0] * s; cy += tmu[k * 3 + 1] * s; cz += tmu[k * 3 + 2] * s;
        }
        weightW[(size_t)b * J + i] = wsum;
        corrW[((size_t)b * 3 + 0) * J + i] = cx;
        corrW[((size_t)b * 3 + 1) * J + i] = cy;
        corrW[((size_t)b * 3 + 2) * J + i] = cz;
        outCorrT[((size_t)b * J + i) * 3 + 0] = cx;
        outCorrT[((size_t)b * J + i) * 3 + 1] = cy;
        outCorrT[((size_t)b * J + i) * 3 + 2] = cz;
    }
}

// ---------------------------------------------------------------------------
// 9) Weighted Kabsch: centroids, 3x3 cross-covariance, Jacobi 3x3 SVD, R & t.
// ---------------------------------------------------------------------------
__global__ __launch_bounds__(128) void k_rigid(const float* __restrict__ muS,
                                               const float* __restrict__ corrW,
                                               const float* __restrict__ weightW,
                                               float* __restrict__ outR,
                                               float* __restrict__ outT)
{
    using namespace cfg;
    __shared__ float red[128];
    __shared__ float acc[16];
    const int b = blockIdx.x, i = threadIdx.x;
    const float w = weightW[(size_t)b * J + i];
    float s[3], cc[3];
    for (int d = 0; d < 3; ++d) {
        s[d]  = muS[((size_t)b * JP + i) * 3 + d];
        cc[d] = corrW[((size_t)b * 3 + d) * J + i];
    }
    auto reduce = [&](float v, int slot) {
        red[i] = v; __syncthreads();
        for (int st = 64; st > 0; st >>= 1) {
            if (i < st) red[i] += red[i + st];
            __syncthreads();
        }
        if (i == 0) acc[slot] = red[0];
        __syncthreads();
    };
    reduce(w, 0);
    for (int d = 0; d < 3; ++d) reduce(s[d] * w, 1 + d);
    for (int d = 0; d < 3; ++d) reduce(cc[d] * w, 4 + d);
    const float sw = acc[0] + EPSV;
    float cs[3], ccm[3];
    for (int d = 0; d < 3; ++d) { cs[d] = acc[1 + d] / sw; ccm[d] = acc[4 + d] / sw; }
    float ds[3], dc[3];
    for (int d = 0; d < 3; ++d) { ds[d] = s[d] - cs[d]; dc[d] = cc[d] - ccm[d]; }
    for (int d = 0; d < 3; ++d)
        for (int e = 0; e < 3; ++e) reduce(ds[d] * w * dc[e], 7 + d * 3 + e);

    if (i == 0) {
        float H[3][3];
        for (int d = 0; d < 3; ++d)
            for (int e = 0; e < 3; ++e) H[d][e] = acc[7 + d * 3 + e];
        float A[3][3];
        for (int d = 0; d < 3; ++d)
            for (int e = 0; e < 3; ++e)
                A[d][e] = H[0][d] * H[0][e] + H[1][d] * H[1][e] + H[2][d] * H[2][e];
        float V[3][3] = {{1, 0, 0}, {0, 1, 0}, {0, 0, 1}};
        const int PP[3] = {0, 0, 1}, QQ[3] = {1, 2, 2};
        for (int sweep = 0; sweep < 12; ++sweep) {
            const int p = PP[sweep % 3], q = QQ[sweep % 3];
            const float apq = A[p][q];
            if (fabsf(apq) > 1e-20f) {
                const float th = 0.5f * atan2f(2.f * apq, A[q][q] - A[p][p]);
                const float c = cosf(th), sn = sinf(th);
                for (int r = 0; r < 3; ++r) {
                    const float arp = A[r][p], arq = A[r][q];
                    A[r][p] = c * arp - sn * arq; A[r][q] = sn * arp + c * arq;
                }
                for (int r = 0; r < 3; ++r) {
                    const float apr = A[p][r], aqr = A[q][r];
                    A[p][r] = c * apr - sn * aqr; A[q][r] = sn * apr + c * aqr;
                }
                for (int r = 0; r < 3; ++r) {
                    const float vrp = V[r][p], vrq = V[r][q];
                    V[r][p] = c * vrp - sn * vrq; V[r][q] = sn * vrp + c * vrq;
                }
            }
        }
        float lam[3] = {A[0][0], A[1][1], A[2][2]};
        for (int a2 = 0; a2 < 2; ++a2)
            for (int b2 = 0; b2 < 2 - a2; ++b2)
                if (lam[b2] < lam[b2 + 1]) {
                    float tl = lam[b2]; lam[b2] = lam[b2 + 1]; lam[b2 + 1] = tl;
                    for (int r = 0; r < 3; ++r) {
                        float tv = V[r][b2]; V[r][b2] = V[r][b2 + 1]; V[r][b2 + 1] = tv;
                    }
                }
        float U[3][3];
        for (int k = 0; k < 3; ++k) {
            const float sg = sqrtf(fmaxf(lam[k], 0.f));
            float uv[3];
            for (int d = 0; d < 3; ++d)
                uv[d] = H[d][0] * V[0][k] + H[d][1] * V[1][k] + H[d][2] * V[2][k];
            if (sg > 1e-9f) {
                for (int d = 0; d < 3; ++d) U[d][k] = uv[d] / sg;
            } else if (k == 2) {
                U[0][2] = U[1][0] * U[2][1] - U[2][0] * U[1][1];
                U[1][2] = U[2][0] * U[0][1] - U[0][0] * U[2][1];
                U[2][2] = U[0][0] * U[1][1] - U[1][0] * U[0][1];
            } else {
                for (int d = 0; d < 3; ++d) U[d][k] = (d == k) ? 1.f : 0.f;
            }
        }
        auto det3 = [](float M[3][3]) {
            return M[0][0] * (M[1][1] * M[2][2] - M[1][2] * M[2][1])
                 - M[0][1] * (M[1][0] * M[2][2] - M[1][2] * M[2][0])
                 + M[0][2] * (M[1][0] * M[2][1] - M[1][1] * M[2][0]);
        };
        const float det = det3(V) * det3(U);
        float R[3][3];
        for (int d = 0; d < 3; ++d)
            for (int e = 0; e < 3; ++e) {
                float r = 0.f;
                for (int k = 0; k < 3; ++k)
                    r += V[d][k] * ((k == 2) ? det : 1.f) * U[e][k];
                R[d][e] = r;
            }
        for (int d = 0; d < 3; ++d) {
            for (int e = 0; e < 3; ++e) outR[(size_t)b * 9 + d * 3 + e] = R[d][e];
            outT[(size_t)b * 3 + d] =
                ccm[d] - (R[d][0] * cs[0] + R[d][1] * cs[1] + R[d][2] * cs[2]);
        }
    }
}

// ---------------------------------------------------------------------------
extern "C" void kernel_launch(void* const* d_in, const int* in_sizes, int n_in,
                              void* d_out, int out_size, void* d_ws, size_t ws_size,
                              hipStream_t stream)
{
    using namespace cfg;
    (void)in_sizes; (void)n_in; (void)out_size; (void)ws_size;
    const float* src   = (const float*)d_in[0];  // (B,3,N)
    const float* tgt   = (const float*)d_in[1];  // (B,3,N)
    const float* sdesc = (const float*)d_in[2];  // (B,N,C)
    const float* tdesc = (const float*)d_in[3];  // (B,N,C)
    const float* slg   = (const float*)d_in[4];  // (B,J,N)
    const float* tlg   = (const float*)d_in[5];  // (B,J,N)
    const float* so    = (const float*)d_in[6];  // (B,N)
    const float* to    = (const float*)d_in[7];  // (B,N)

    float* out = (float*)d_out;                  // [R(288) | t(96) | corrT(12288) | scores(524288)]
    float* outR      = out;
    float* outT      = out + 288;
    float* outCorrT  = out + 384;
    float* outScores = out + 12672;

    char*  w   = (char*)d_ws;
    size_t off = 0;
    auto take = [&](size_t bytes) {
        size_t o = off; off += (bytes + 255) & ~(size_t)255; return o;
    };
    _Float16* WtS = (_Float16*)(w + take((size_t)B * JP * N * 2));
    _Float16* WtT = (_Float16*)(w + take((size_t)B * JP * N * 2));
    _Float16* dTS = (_Float16*)(w + take((size_t)B * C * N * 2));
    _Float16* dTT = (_Float16*)(w + take((size_t)B * C * N * 2));
    float* denS = (float*)(w + take((size_t)B * JP * 4));
    float* piS  = (float*)(w + take((size_t)B * JP * 4));
    float* denT = (float*)(w + take((size_t)B * JP * 4));
    float* piT  = (float*)(w + take((size_t)B * JP * 4));
    float* muS  = (float*)(w + take((size_t)B * JP * 3 * 4));
    float* muT  = (float*)(w + take((size_t)B * JP * 3 * 4));
    float* PS   = (float*)(w + take((size_t)B * JP * C * 4));
    float* PT   = (float*)(w + take((size_t)B * JP * C * 4));
    _Float16* NS = (_Float16*)(w + take((size_t)B * JP * C * 2));
    _Float16* NT = (_Float16*)(w + take((size_t)B * JP * C * 2));
    float* cost  = (float*)(w + take((size_t)B * CSTR * 4));
    float* wW    = (float*)(w + take((size_t)B * J * 4));
    float* corrW = (float*)(w + take((size_t)B * 3 * J * 4));

    const dim3 gSm(N / 256, B), gJB(JP, B);
    k_softmax_w<<<gSm, 256, 0, stream>>>(slg, so, WtS);
    k_softmax_w<<<gSm, 256, 0, stream>>>(tlg, to, WtT);
    k_colsum<<<gJB, 256, 0, stream>>>(WtS, denS, piS);
    k_colsum<<<gJB, 256, 0, stream>>>(WtT, denT, piT);
    k_mu<<<gJB, 256, 0, stream>>>(WtS, src, denS, muS);
    k_mu<<<gJB, 256, 0, stream>>>(WtT, tgt, denT, muT);

    const dim3 gTr(N / 32, C / 32, B), bTr(32, 8);
    k_transpose<<<gTr, bTr, 0, stream>>>(sdesc, dTS);
    k_transpose<<<gTr, bTr, 0, stream>>>(tdesc, dTT);

    const int blkDesc = (B * (JP / 16) * (C / (16 * CB))) / 8;   // 144
    k_gemm_descmu<<<blkDesc, 256, 0, stream>>>(WtS, dTS, denS, PS);
    k_gemm_descmu<<<blkDesc, 256, 0, stream>>>(WtT, dTT, denT, PT);

    k_swap_norm<<<gJB, 256, 0, stream>>>(PS, PT, NS);
    k_swap_norm<<<gJB, 256, 0, stream>>>(PT, PS, NT);

    const int blkSim = (B * (JP / 16) * (JP / 16) + 7) / 8;      // 324
    k_gemm_sim<<<blkSim, 256, 0, stream>>>(NS, NT, cost);

    const size_t sinkLds = (size_t)(CSTR + 4 * JO + J * 3) * sizeof(float);
    k_sinkhorn<<<B, 256, sinkLds, stream>>>(cost, piS, piT, muT,
                                            outScores, outCorrT, wW, corrW);
    k_rigid<<<B, 128, 0, stream>>>(muS, corrW, wW, outR, outT);
}
